// DCN_62766652064420
// MI455X (gfx1250) — matrix operand order repaired and checked by reading, not compiled, not run on hardware
//
#include <hip/hip_runtime.h>
#include <hip/hip_fp16.h>

typedef __attribute__((ext_vector_type(16))) _Float16 v16h;
typedef __attribute__((ext_vector_type(8)))  float    v8f;

#define BATCH 16
#define CIN   128
#define HH    56
#define WW    56
#define OUTC  256
#define HW    (HH*WW)        // 3136
#define KDIM  (CIN*9)        // 1152
#define BM    64
#define BN    256
#define BK    32
#define NCHUNK (KDIM/BK)     // 36

struct Meta { int i00, i01, i10, i11; float w00, w01, w10, w11; };

union V16 { uint4 u4[2]; v16h h; };

typedef __attribute__((address_space(3))) _Float16 lds_f16_t;

// LDS byte offset (workgroup-relative) of a generic shared pointer.
__device__ __forceinline__ uint32_t lds_offset(const _Float16* p) {
  return (uint32_t)(uintptr_t)(lds_f16_t*)(p);
}

// Async copy of one 64B B-row into LDS (ASYNCcnt-tracked; offset adds to BOTH addrs).
__device__ __forceinline__ void async_copy_row64(uint32_t lds_off, const _Float16* gsrc) {
  uint64_t ga = (uint64_t)(uintptr_t)gsrc;
  asm volatile(
      "global_load_async_to_lds_b128 %0, %1, off\n\t"
      "global_load_async_to_lds_b128 %0, %1, off offset:16\n\t"
      "global_load_async_to_lds_b128 %0, %1, off offset:32\n\t"
      "global_load_async_to_lds_b128 %0, %1, off offset:48"
      :: "v"(lds_off), "v"(ga) : "memory");
}

__device__ __forceinline__ void wait_async0() {
  asm volatile("s_wait_asynccnt 0x0" ::: "memory");
}

// Issue the 4-corner gathers (8 channels) for one A element row; no waiting here.
__device__ __forceinline__ void gather_issue(const float* fc, const Meta& md,
                                             float* g00, float* g01,
                                             float* g10, float* g11) {
  #pragma unroll
  for (int cc = 0; cc < 8; ++cc) {
    const float* f = fc + (size_t)cc * HW;
    g00[cc] = f[md.i00];
    g01[cc] = f[md.i01];
    g10[cc] = f[md.i10];
    g11[cc] = f[md.i11];
  }
}

// Weighted sum + f16 convert + LDS store (this is where loadcnt is consumed).
__device__ __forceinline__ void gather_commit(const Meta& md,
                                              const float* g00, const float* g01,
                                              const float* g10, const float* g11,
                                              _Float16* dst) {
  #pragma unroll
  for (int cc = 0; cc < 8; ++cc) {
    float v = md.w00 * g00[cc] + md.w01 * g01[cc]
            + md.w10 * g10[cc] + md.w11 * g11[cc];
    dst[cc] = (_Float16)v;
  }
}

// Pack weight [OUTC][CIN][3][3] fp32 -> Wt[oc][kIdx] f16, kIdx = p*128 + c
__global__ __launch_bounds__(256) void dcn_weight_pack(const float* __restrict__ w,
                                                       _Float16* __restrict__ wt) {
  int t = blockIdx.x * 256 + threadIdx.x;       // 0 .. OUTC*KDIM-1
  int n = t / KDIM;
  int kidx = t - n * KDIM;
  int p = kidx >> 7;                            // kernel tap 0..8
  int c = kidx & 127;                           // channel
  wt[t] = (_Float16)w[(n * CIN + c) * 9 + p];
}

__global__ __launch_bounds__(256) void dcn_fused_gemm(const float* __restrict__ feature,
                                                      const float* __restrict__ offset,
                                                      const _Float16* __restrict__ wt,
                                                      float* __restrict__ out) {
  __shared__ Meta     meta[BM * 9];        // 18432 B
  __shared__ _Float16 sA[2][BM * 40];      // 2 x 5120 B, padded stride 40 halves
  __shared__ _Float16 sB[2][BN * 40];      // 2 x 20480 B

  const int t    = threadIdx.x;
  const int m0   = blockIdx.x * BM;        // first output pixel of tile
  const int bIdx = m0 / HW;                // uniform: 64 | 3136
  const int s0   = m0 - bIdx * HW;         // spatial base (yo*56+xo)

  // ---- Phase 0: bilinear sampling metadata (once per tile) ----
  for (int e = t; e < BM * 9; e += 256) {
    int r  = e / 9;
    int p  = e - r * 9;
    int s  = s0 + r;
    int yo = s / WW;
    int xo = s - yo * WW;
    // Reference transposes the spatial index into offset: [b, ch, xo, yo]
    const float* ob = offset + ((size_t)bIdx * 18 + 2 * p) * HW + xo * WW + yo;
    float ox = ob[0];
    float oy = ob[HW];
    float gx = ox * 27.5f + (float)xo;     // anchor math collapses to this
    float gy = oy * 27.5f + (float)yo;
    float x0f = floorf(gx), y0f = floorf(gy);
    float wx1 = gx - x0f, wy1 = gy - y0f;
    float wx0 = 1.0f - wx1, wy0 = 1.0f - wy1;
    int x0 = (int)x0f, y0 = (int)y0f;
    int x1 = x0 + 1,   y1 = y0 + 1;
    bool vx0 = (x0 >= 0) && (x0 < WW);
    bool vx1 = (x1 >= 0) && (x1 < WW);
    bool vy0 = (y0 >= 0) && (y0 < HH);
    bool vy1 = (y1 >= 0) && (y1 < HH);
    int x0c = min(max(x0, 0), WW - 1), x1c = min(max(x1, 0), WW - 1);
    int y0c = min(max(y0, 0), HH - 1), y1c = min(max(y1, 0), HH - 1);
    Meta md;
    md.i00 = y0c * WW + x0c;  md.i01 = y0c * WW + x1c;
    md.i10 = y1c * WW + x0c;  md.i11 = y1c * WW + x1c;
    md.w00 = (vx0 && vy0) ? wx0 * wy0 : 0.0f;
    md.w01 = (vx1 && vy0) ? wx1 * wy0 : 0.0f;
    md.w10 = (vx0 && vy1) ? wx0 * wy1 : 0.0f;
    md.w11 = (vx1 && vy1) ? wx1 * wy1 : 0.0f;
    meta[e] = md;
  }
  __syncthreads();

  const int lane = t & 31;
  const int wv   = t >> 5;        // wave 0..7
  const int wm   = wv & 3;        // M block (16 rows each)
  const int wn   = wv >> 2;       // N half (128 cols each)
  const int l16  = lane & 15;
  const int g    = lane >> 4;

  v8f acc[8] = {};                // wave tile: 16 x 128 (8 frags of 16x16 f32)

  const int rA = t & 63;          // pixel row this thread fills in sA
  const int cg = t >> 6;          // channel subgroup 0..3 (8 channels each)
  const float* fbase = feature + (size_t)bIdx * CIN * HW;
  const _Float16* wrow = wt + (size_t)t * KDIM;   // B-fill: one oc row per thread

  float g00[8], g01[8], g10[8], g11[8];
  Meta md;

  // ---- Prologue: stage chunk 0 into buffer 0 ----
  {
    md = meta[rA * 9 + 0];
    gather_issue(fbase + (size_t)(cg * 8) * HW, md, g00, g01, g10, g11);
    async_copy_row64(lds_offset(&sB[0][t * 40]), wrow);
    gather_commit(md, g00, g01, g10, g11, &sA[0][rA * 40 + cg * 8]);
    wait_async0();
  }
  __syncthreads();

  for (int chunk = 0; chunk < NCHUNK; ++chunk) {
    const int cur = chunk & 1;
    const int nxt = cur ^ 1;
    const bool more = (chunk + 1) < NCHUNK;

    // ---- Issue next chunk's gathers + async B copy (no waits) ----
    if (more) {
      int kn  = (chunk + 1) * BK;
      int pn  = kn >> 7;
      int c0n = kn & 127;
      md = meta[rA * 9 + pn];
      gather_issue(fbase + (size_t)(c0n + cg * 8) * HW, md, g00, g01, g10, g11);
      async_copy_row64(lds_offset(&sB[nxt][t * 40]), wrow + kn);
    }

    // ---- Consume current chunk: frag loads + 8 WMMAs ----
    {
      V16 a;
      const _Float16* ap = &sA[cur][(wm * 16 + l16) * 40 + g * 8];
      a.u4[0] = *(const uint4*)ap;          // halves K = 8g .. 8g+7
      a.u4[1] = *(const uint4*)(ap + 16);   // halves K = 8g+16 .. 8g+23

      #pragma unroll
      for (int f = 0; f < 8; ++f) {
        V16 b;
        const _Float16* bp = &sB[cur][(wn * 128 + f * 16 + l16) * 40 + g * 8];
        b.u4[0] = *(const uint4*)bp;
        b.u4[1] = *(const uint4*)(bp + 16);
        acc[f] = __builtin_amdgcn_wmma_f32_16x16x32_f16(
            /*neg_a=*/false, a.h, /*neg_b=*/false, b.h,
            /*c_mod=*/(short)0, acc[f], /*reuse_a=*/false, /*reuse_b=*/false);
      }
    }

    // ---- Commit next chunk's A tile; drain async B; barrier ----
    if (more) {
      gather_commit(md, g00, g01, g10, g11, &sA[nxt][rA * 40 + cg * 8]);
      wait_async0();
    }
    __syncthreads();
  }

  // ---- Store: per lane, VGPR r covers M = r + 8g -> 8 consecutive pixels ----
  const int sBase = s0 + wm * 16 + g * 8;
  #pragma unroll
  for (int f = 0; f < 8; ++f) {
    int oc = wn * 128 + f * 16 + l16;
    float* o = out + ((size_t)bIdx * OUTC + oc) * HW + sBase;
    float4 lo = make_float4(acc[f][0], acc[f][1], acc[f][2], acc[f][3]);
    float4 hi = make_float4(acc[f][4], acc[f][5], acc[f][6], acc[f][7]);
    *(float4*)(o)     = lo;
    *(float4*)(o + 4) = hi;
  }
}

extern "C" void kernel_launch(void* const* d_in, const int* in_sizes, int n_in,
                              void* d_out, int out_size, void* d_ws, size_t ws_size,
                              hipStream_t stream) {
  (void)in_sizes; (void)n_in; (void)out_size; (void)ws_size;
  const float* feature = (const float*)d_in[0];   // [16,128,56,56]
  const float* offset  = (const float*)d_in[1];   // [16,18,56,56]
  const float* weight  = (const float*)d_in[2];   // [256,128,3,3]
  float* out = (float*)d_out;                     // [16,256,56,56]
  _Float16* wt = (_Float16*)d_ws;                 // 256*1152 f16 = 576 KiB

  dcn_weight_pack<<<(OUTC * KDIM) / 256, 256, 0, stream>>>(weight, wt);
  dcn_fused_gemm<<<(BATCH * HW) / BM, 256, 0, stream>>>(feature, offset, wt, out);
}
